// BaseTransformerLayer_68358699483732
// MI455X (gfx1250) — matrix operand order repaired and checked by reading, not compiled, not run on hardware
//
#include <hip/hip_runtime.h>

#define HNUM   4
#define DH     32
#define OH     128
#define AIN    256
#define SCALE_F 0.08838834764831845f   // 1/sqrt(128)

typedef _Float16 half_t;
typedef __attribute__((ext_vector_type(16))) _Float16 v16h;
typedef __attribute__((ext_vector_type(8)))  float    v8f;

__device__ __forceinline__ v8f wmma16(v16h a, v16h b, v8f c) {
  // D = A(16x32 f16) * B(32x16 f16) + C(16x16 f32)
  return __builtin_amdgcn_wmma_f32_16x16x32_f16(false, a, false, b, (short)0, c,
                                                false, false);
}

// A fragment: 16-bit A 16x32 layout (ISA 7.12.2). lane<16 -> K {0..7,16..23},
// lane>=16 -> K {8..15,24..31}; row = row0 + (lane&15). f32 -> f16 on load.
__device__ __forceinline__ v16h load_a_frag(const float* __restrict__ A, int lda,
                                            int row0, int kb, int lane) {
  const int m  = lane & 15;
  const int kh = lane >> 4;
  const float* ap = A + (size_t)(row0 + m) * lda + kb;
  v16h a;
#pragma unroll
  for (int v = 0; v < 8; ++v) {
    const int k = ((v < 4) ? (2 * v) : (16 + 2 * (v - 4))) + 8 * kh;
    const float2 p = *(const float2*)(ap + k);
    a[2 * v]     = (_Float16)p.x;
    a[2 * v + 1] = (_Float16)p.y;
  }
  return a;
}

// B fragments pre-swizzled: tile = (kb/32)*nct + (col0/16), 512 halves/tile,
// lane-major groups of 16 contiguous halves -> one 32B load per fragment.
__device__ __forceinline__ v16h load_b_frag(const half_t* __restrict__ Wsw, int nct,
                                            int kb, int col0, int lane) {
  const size_t off = (((size_t)(kb >> 5) * nct + (col0 >> 4)) << 9) + ((size_t)lane << 4);
  return *(const v16h*)(Wsw + off);
}

// Swizzle a (K x 128) f32 weight into fragment-native f16 layout.
// b[j] of lane l = W[kb + 8*(l>>4) + (j&7) + 16*(j>>3)][col0 + (l&15)]
__global__ __launch_bounds__(256) void swizzle_w_kernel(const float* __restrict__ W,
                                                        half_t* __restrict__ out,
                                                        int K, int ncols) {
  const int idx = blockIdx.x * blockDim.x + threadIdx.x;
  if (idx >= K * ncols) return;
  const int j    = idx & 15;
  const int lane = (idx >> 4) & 31;
  const int tile = idx >> 9;
  const int nct  = ncols >> 4;
  const int kb   = (tile / nct) * 32;
  const int ct   = (tile % nct) * 16;
  const int k    = kb + 8 * (lane >> 4) + (j & 7) + ((j >> 3) << 4);
  const int col  = ct + (lane & 15);
  out[idx] = (half_t)W[(size_t)k * ncols + col];
}

// q99/k99/v99 = x99 @ {Waq,Wak,Wav} + bias. One A-fragment feeds 3 WMMAs.
__global__ __launch_bounds__(256) void qkv_gemm_kernel(
    const float* __restrict__ x99,
    const half_t* __restrict__ Wq, const half_t* __restrict__ Wk,
    const half_t* __restrict__ Wv,
    const float* __restrict__ bq, const float* __restrict__ bk,
    const float* __restrict__ bv,
    float* __restrict__ q, float* __restrict__ k, float* __restrict__ v) {
  const int lane = threadIdx.x & 31;
  const int wave = threadIdx.x >> 5;
  const int row0 = blockIdx.x << 4;
  const int col0 = wave << 4;
  v8f aq = {}, ak = {}, av = {};
#pragma unroll
  for (int kb = 0; kb < AIN; kb += 32) {
    v16h a = load_a_frag(x99, AIN, row0, kb, lane);
    aq = wmma16(a, load_b_frag(Wq, 8, kb, col0, lane), aq);
    ak = wmma16(a, load_b_frag(Wk, 8, kb, col0, lane), ak);
    av = wmma16(a, load_b_frag(Wv, 8, kb, col0, lane), av);
  }
  const int col = col0 + (lane & 15);
  const int mh  = (lane >> 4) << 3;
  const float bqv = bq[col], bkv = bk[col], bvv = bv[col];
#pragma unroll
  for (int r = 0; r < 8; ++r) {
    const size_t o = (size_t)(row0 + mh + r) * OH + col;
    q[o] = aq[r] + bqv;
    k[o] = ak[r] + bkv;
    v[o] = av[r] + bvv;
  }
}

// One wave per edge; lane = d within head; per-head shuffle reduction,
// then native f32 atomics into wv (N x 128) and z (N x 4).
__global__ __launch_bounds__(256) void edge_attn_kernel(
    const float* __restrict__ q, const float* __restrict__ k,
    const float* __restrict__ v, const float* __restrict__ feat,
    const int* __restrict__ src, const int* __restrict__ dst,
    float* __restrict__ wv, float* __restrict__ z, int E) {
  const int e    = (int)((blockIdx.x * (size_t)blockDim.x + threadIdx.x) >> 5);
  const int lane = threadIdx.x & 31;
  if (e >= E) return;
  const int s = src[e];
  const int d = dst[e];
  const float f = feat[2 * (size_t)e];
  const float* kp = k + (size_t)s * OH;
  const float* qp = q + (size_t)d * OH;
  const float* vp = v + (size_t)s * OH;
  float sh[HNUM];
#pragma unroll
  for (int h = 0; h < HNUM; ++h) {
    float p = kp[h * DH + lane] * qp[h * DH + lane];
#pragma unroll
    for (int off = 16; off >= 1; off >>= 1) p += __shfl_xor(p, off, 32);
    float sc = p * f * SCALE_F;
    sc = fminf(5.0f, fmaxf(-5.0f, sc));
    sh[h] = expf(sc);
  }
#pragma unroll
  for (int h = 0; h < HNUM; ++h)
    unsafeAtomicAdd(&wv[(size_t)d * OH + h * DH + lane], vp[h * DH + lane] * sh[h]);
  if (lane == 0) {
#pragma unroll
    for (int h = 0; h < HNUM; ++h) unsafeAtomicAdd(&z[(size_t)d * HNUM + h], sh[h]);
  }
}

// ah = (wv/(z+1)) @ Wao + bao + x99 @ Waffn + baffn ; out = ah + LN(ah)*g + b
__global__ __launch_bounds__(256) void out_kernel(
    const float* __restrict__ wv, const float* __restrict__ z,
    const float* __restrict__ x99,
    const half_t* __restrict__ Wao, const half_t* __restrict__ Waffn,
    const float* __restrict__ bao, const float* __restrict__ baffn,
    const float* __restrict__ g, const float* __restrict__ b,
    float* __restrict__ out) {
  __shared__ float tile[16][OH];
  const int lane = threadIdx.x & 31;
  const int wave = threadIdx.x >> 5;
  const int row0 = blockIdx.x << 4;
  const int col0 = wave << 4;
  const int m  = lane & 15;
  const int kh = lane >> 4;
  v8f acc = {};
  // G1: (wv / (z+1)) @ Wao, K = 128; whole k-block lives in one head.
#pragma unroll
  for (int kb = 0; kb < OH; kb += 32) {
    const float* wp = wv + (size_t)(row0 + m) * OH + kb;
    const float zz = 1.0f / (z[(size_t)(row0 + m) * HNUM + (kb >> 5)] + 1.0f);
    v16h a;
#pragma unroll
    for (int vv = 0; vv < 8; ++vv) {
      const int kk = ((vv < 4) ? (2 * vv) : (16 + 2 * (vv - 4))) + 8 * kh;
      const float2 p = *(const float2*)(wp + kk);
      a[2 * vv]     = (_Float16)(p.x * zz);
      a[2 * vv + 1] = (_Float16)(p.y * zz);
    }
    acc = wmma16(a, load_b_frag(Wao, 8, kb, col0, lane), acc);
  }
  // G2: x99 @ Waffn, K = 256, accumulated into the same tile.
#pragma unroll
  for (int kb = 0; kb < AIN; kb += 32) {
    v16h a = load_a_frag(x99, AIN, row0, kb, lane);
    acc = wmma16(a, load_b_frag(Waffn, 8, kb, col0, lane), acc);
  }
  const int col = col0 + (lane & 15);
  const int mh  = (lane >> 4) << 3;
  const float bias = bao[col] + baffn[col];
#pragma unroll
  for (int r = 0; r < 8; ++r) tile[mh + r][col] = acc[r] + bias;
  __syncthreads();
  // Residual + LayerNorm: each wave owns 2 rows, 4 elems/lane.
#pragma unroll
  for (int rr = 0; rr < 2; ++rr) {
    const int r = (wave << 1) + rr;
    float vals[4], s = 0.0f;
#pragma unroll
    for (int j = 0; j < 4; ++j) { vals[j] = tile[r][lane + 32 * j]; s += vals[j]; }
#pragma unroll
    for (int off = 16; off >= 1; off >>= 1) s += __shfl_xor(s, off, 32);
    const float mean = s * (1.0f / 128.0f);
    float vs = 0.0f;
#pragma unroll
    for (int j = 0; j < 4; ++j) { const float d0 = vals[j] - mean; vs += d0 * d0; }
#pragma unroll
    for (int off = 16; off >= 1; off >>= 1) vs += __shfl_xor(vs, off, 32);
    const float rs = rsqrtf(vs * (1.0f / 128.0f) + 1e-5f);
    const size_t ro = (size_t)(row0 + r) * OH;
#pragma unroll
    for (int j = 0; j < 4; ++j) {
      const int c = lane + 32 * j;
      const float ah = vals[j];
      out[ro + c] = ah + (ah - mean) * rs * g[c] + b[c];
    }
  }
}

extern "C" void kernel_launch(void* const* d_in, const int* in_sizes, int n_in,
                              void* d_out, int out_size, void* d_ws, size_t ws_size,
                              hipStream_t stream) {
  // setup_inputs() order
  const float* x99     = (const float*)d_in[2];
  const float* feats[2]= {(const float*)d_in[3], (const float*)d_in[4]};
  const float* Waq     = (const float*)d_in[10];
  const float* Wak     = (const float*)d_in[11];
  const float* Wav     = (const float*)d_in[12];
  const float* Wao     = (const float*)d_in[13];
  const float* Waffn   = (const float*)d_in[14];
  const float* baq     = (const float*)d_in[20];
  const float* bak     = (const float*)d_in[21];
  const float* bav     = (const float*)d_in[22];
  const float* bao     = (const float*)d_in[23];
  const float* baffn   = (const float*)d_in[24];
  const float* aln_b   = (const float*)d_in[26];
  const float* aln_g   = (const float*)d_in[28];
  const int* srcs[2]   = {(const int*)d_in[29], (const int*)d_in[31]};
  const int* dsts[2]   = {(const int*)d_in[30], (const int*)d_in[32]};

  const int N = in_sizes[2] / AIN;     // x99 is N x 256 (N = 50000, mult of 16)
  const int E = in_sizes[29];          // 800000
  const size_t NF = (size_t)N * OH;

  float* ws  = (float*)d_ws;
  float* q99 = ws;
  float* k99 = q99 + NF;
  float* v99 = k99 + NF;
  float* wvb = v99 + NF;
  float* zb  = wvb + NF;
  uintptr_t hp = (uintptr_t)(zb + (size_t)N * HNUM);
  hp = (hp + 63) & ~(uintptr_t)63;
  half_t* Waqh   = (half_t*)hp;         // 256*128
  half_t* Wakh   = Waqh + 32768;
  half_t* Wavh   = Wakh + 32768;
  half_t* Waffnh = Wavh + 32768;
  half_t* Waoh   = Waffnh + 32768;      // 128*128
  (void)ws_size; (void)n_in; (void)out_size;

  // One-time (per call) weight swizzle f32 -> fragment-native f16.
  swizzle_w_kernel<<<(32768 + 255) / 256, 256, 0, stream>>>(Waq,   Waqh,   256, 128);
  swizzle_w_kernel<<<(32768 + 255) / 256, 256, 0, stream>>>(Wak,   Wakh,   256, 128);
  swizzle_w_kernel<<<(32768 + 255) / 256, 256, 0, stream>>>(Wav,   Wavh,   256, 128);
  swizzle_w_kernel<<<(32768 + 255) / 256, 256, 0, stream>>>(Waffn, Waffnh, 256, 128);
  swizzle_w_kernel<<<(16384 + 255) / 256, 256, 0, stream>>>(Wao,   Waoh,   128, 128);

  // q99/k99/v99 projections (shared by both edge sets).
  qkv_gemm_kernel<<<N / 16, 256, 0, stream>>>(x99, Waqh, Wakh, Wavh,
                                              baq, bak, bav, q99, k99, v99);

  for (int i = 0; i < 2; ++i) {
    hipMemsetAsync(wvb, 0, NF * sizeof(float), stream);
    hipMemsetAsync(zb,  0, (size_t)N * HNUM * sizeof(float), stream);
    edge_attn_kernel<<<(E + 7) / 8, 256, 0, stream>>>(q99, k99, v99, feats[i],
                                                      srcs[i], dsts[i], wvb, zb, E);
    out_kernel<<<N / 16, 256, 0, stream>>>(wvb, zb, x99, Waoh, Waffnh,
                                           bao, baffn, aln_g, aln_b,
                                           (float*)d_out + (size_t)i * NF);
  }
}
